// PR_GNN_20564303414148
// MI455X (gfx1250) — matrix operand order repaired and checked by reading, not compiled, not run on hardware
//
#include <hip/hip_runtime.h>
#include <stdint.h>

// ---------------------------------------------------------------------------
// GNN (GAT + top-k pool) pipeline for gfx1250 (MI455X), wave32.
// WMMA f16->f32 for all GEMMs; TDM (tensor_load_to_lds) stages x into LDS.
// ---------------------------------------------------------------------------

#define NODE  112
#define L1D   256
#define L2D   128
#define L3D   64
#define KP1   56
#define KP2   28
#define BATCH 2048

typedef __attribute__((ext_vector_type(16))) _Float16 v16h;
typedef __attribute__((ext_vector_type(8)))  float    v8f;
typedef __attribute__((ext_vector_type(8)))  uint32_t v8u;
typedef __attribute__((ext_vector_type(4)))  uint32_t u32x4;
typedef __attribute__((ext_vector_type(8)))  int      i32x8;
typedef __attribute__((ext_vector_type(4)))  int      i32x4;

union FragU { v8u u; v16h h; };

// Workspace byte offsets (constants precomputed once, L2-resident afterwards)
#define WS_HF32T 0                         // float  [256][112]  (H^T, f32)   114688 B
#define WS_HF16T 114688                    // f16    [256][128]  (H^T, k-pad)  65536 B
#define WS_EXPE  180224                    // float  [112][112]  exp(relu(E))  50176 B
#define WS_WP2H  230400                    // f16    [128][256]  Wp2 (=B^T)    65536 B
#define WS_TOTAL 295936

// LDS arena byte offsets (phase-overlapped; peak 155,264 B)
#define OFF_X    0          // float [112][112]           (phase A)
#define OFF_H2   0          // float [64][128]            (phase B, x dead)
#define OFF_H2T  32768      // f16   [128][64]            (phase B)
#define OFF_AT2  49152      // f16   [64][64]             (phase B)
#define OFF_H1   50176      // f16   [112][256]           (phase A)
#define OFF_O2   57344      // float [64][128]            (phase B, h1 dead)
#define OFF_A    107520     // f16   attA[112][128] then hpA[64][256]
#define OFF_A1   140288     // float [56][56]
#define OFF_SC   152832     // float [112]
#define OFF_IDX  153280     // int   [112]
#define OFF_VAL  153728     // float [64]
#define OFF_S2   153984     // float [64]
#define OFF_T2   154240     // float [64]
#define OFF_MEAN 154496     // float [128]
#define OFF_Y1   155008     // float [64]
#define LDS_SIZE 155264

// ---- TDM: DMA a 2-D f32 tile (rows x cols, row stride = cols) into LDS -----
// D# packing per CDNA5 ISA §8.3 (group 0) / §8.4 (group 1). Groups 2/3 zero
// (tile_dim2/3/4 = 0 => unused; plain 2-D tile). Tracked by TENSORcnt.
__device__ __forceinline__ void tdm_load2d_f32_to_lds(const float* gptr,
                                                      uint32_t lds_byte_addr,
                                                      uint32_t rows,
                                                      uint32_t cols) {
  const uint64_t ga = (uint64_t)(uintptr_t)gptr;
  u32x4 g0;
  g0[0] = 1u;                                   // count=1, user descriptor
  g0[1] = lds_byte_addr;                        // lds_addr [63:32]
  g0[2] = (uint32_t)ga;                         // global_addr[31:0]
  g0[3] = (uint32_t)((ga >> 32) & 0x01FFFFFFu)  // global_addr[56:32]
        | (2u << 30);                           // type=2 ("image")
  i32x8 g1;
  g1[0] = (int)(2u << 16);                      // data_size=2 (4 bytes)
  g1[1] = (int)((cols & 0xFFFFu) << 16);        // tensor_dim0[15:0] @ [31:16]
  g1[2] = (int)(((cols >> 16) & 0xFFFFu)        // tensor_dim0[31:16]
        | ((rows & 0xFFFFu) << 16));            // tensor_dim1[15:0]
  g1[3] = (int)(((rows >> 16) & 0xFFFFu)        // tensor_dim1[31:16]
        | ((cols & 0xFFFFu) << 16));            // tile_dim0
  g1[4] = (int)(rows & 0xFFFFu);                // tile_dim1 (tile_dim2=0)
  g1[5] = (int)cols;                            // tensor_dim0_stride[31:0]
  g1[6] = 0;                                    // stride0 hi / stride1 lo
  g1[7] = 0;                                    // stride1 hi
  i32x4 z4 = {0, 0, 0, 0};
#if __clang_major__ >= 23
  i32x8 z8 = {0, 0, 0, 0, 0, 0, 0, 0};
  __builtin_amdgcn_tensor_load_to_lds(g0, g1, z4, z4, z8, 0);
#else
  __builtin_amdgcn_tensor_load_to_lds(g0, g1, z4, z4, 0);
#endif
}

// ---- WMMA fragment loaders (wave32; A and B use the same contiguous k order) --------------------
// A: 16x32 f16, row-major source [rows][Kpad] halves.
//   lane l (row = l&15): k = 16*(l>>4) + e, e = 0..15
__device__ __forceinline__ v16h load_frag_A(const _Float16* base, int strideU32,
                                            int mt, int kk, int lane) {
  const uint32_t* p = (const uint32_t*)base;
  const int row = mt * 16 + (lane & 15);
  const int kb  = kk * 16 + ((lane >> 4) << 3);   // u32 units
  const uint32_t* r = p + row * strideU32 + kb;
  FragU f;
#pragma unroll
  for (int j = 0; j < 4; ++j) f.u[j] = r[j];
#pragma unroll
  for (int j = 0; j < 4; ++j) f.u[4 + j] = r[4 + j];
  return f.h;
}

// B: 32x16 f16 loaded from B^T stored [N][Kpad] halves (contiguous per lane).
//   lane l (n = l&15): k = 16*(l>>4) + e, e = 0..15
__device__ __forceinline__ v16h load_frag_Bt(const _Float16* base, int strideU32,
                                             int nt, int kk, int lane) {
  const uint32_t* p = (const uint32_t*)base;
  const int n  = nt * 16 + (lane & 15);
  const int kb = kk * 16 + ((lane >> 4) << 3);    // u32 units
  const uint32_t* r = p + n * strideU32 + kb;
  FragU f;
#pragma unroll
  for (int j = 0; j < 8; ++j) f.u[j] = r[j];
  return f.h;
}

// jax.lax.top_k rank: #{j: s[j]>s[i]} + #{j<i: s[j]==s[i]}
__device__ __forceinline__ int topk_rank(const float* s, int n, int i) {
  const float si = s[i];
  int r = 0;
  for (int j = 0; j < n; ++j) {
    const float sj = s[j];
    r += (sj > si) || (sj == si && j < i);
  }
  return r;
}

__device__ __forceinline__ float sigmoidf_(float x) {
  return 1.0f / (1.0f + __expf(-x));
}

// ---------------------------------------------------------------------------
// Setup kernel: batch-invariant constants (H = Wp1^T + bp1, E-logits, Wp2 f16)
// ---------------------------------------------------------------------------
__global__ __launch_bounds__(256) void gnn_setup(
    const float* __restrict__ Wp1, const float* __restrict__ bp1,
    const float* __restrict__ wa1, const float* __restrict__ wb1,
    const float* __restrict__ ba1, const float* __restrict__ Wp2,
    float* __restrict__ Hf32t, _Float16* __restrict__ Hf16t,
    float* __restrict__ expE, _Float16* __restrict__ Wp2h) {
  const int tid = threadIdx.x;
  __shared__ float s1[NODE], t1[NODE];

  // H^T[o][m] = Wp1[o][m] + bp1[o]   (f32 + k-padded f16 copy)
  for (int e = tid; e < L1D * NODE; e += 256) {
    const int o = e / NODE, m = e % NODE;
    const float v = Wp1[e] + bp1[o];
    Hf32t[e] = v;
    Hf16t[o * 128 + m] = (_Float16)v;
  }
  for (int e = tid; e < L1D * 16; e += 256) {
    const int o = e >> 4, mm = NODE + (e & 15);
    Hf16t[o * 128 + mm] = (_Float16)0.0f;
  }
  // s1/t1 directly from inputs (no cross-thread global dependency)
  if (tid < NODE) {
    float s = 0.0f, t = 0.0f;
    for (int o = 0; o < L1D; ++o) {
      const float v = Wp1[o * NODE + tid] + bp1[o];
      s += v * wa1[o];
      t += v * wb1[o];
    }
    s1[tid] = s;
    t1[tid] = t;
  }
  __syncthreads();
  const float b1 = ba1[0];
  for (int e = tid; e < NODE * NODE; e += 256) {
    const int i = e / NODE, j = e % NODE;
    expE[e] = __expf(fmaxf(s1[i] + t1[j] + b1, 0.0f));
  }
  for (int e = tid; e < L2D * L1D; e += 256) Wp2h[e] = (_Float16)Wp2[e];
}

// ---------------------------------------------------------------------------
// Main kernel: one workgroup (8 wave32) per batch element, full pipeline in LDS
// ---------------------------------------------------------------------------
__global__ __launch_bounds__(256) void gnn_main(
    const float* __restrict__ x,
    const float* __restrict__ expE, const _Float16* __restrict__ Hf16t,
    const float* __restrict__ Hf32t,
    const float* __restrict__ Wpool1, const float* __restrict__ bpool1,
    const _Float16* __restrict__ Wp2h, const float* __restrict__ bp2,
    const float* __restrict__ wa2, const float* __restrict__ wb2,
    const float* __restrict__ ba2,
    const float* __restrict__ Wpool2, const float* __restrict__ bpool2,
    const float* __restrict__ Wfc1, const float* __restrict__ bfc1,
    const float* __restrict__ Wfc2, const float* __restrict__ bfc2,
    float* __restrict__ out) {
  __shared__ __align__(16) unsigned char smem[LDS_SIZE];

  float*    sX    = (float*)(smem + OFF_X);      // [112][112]
  _Float16* sH1   = (_Float16*)(smem + OFF_H1);  // [112][256]
  _Float16* sA    = (_Float16*)(smem + OFF_A);   // attA [112][128] / hpA [64][256]
  float*    sA1   = (float*)(smem + OFF_A1);     // [56][56]
  float*    sSc   = (float*)(smem + OFF_SC);
  int*      sIdx  = (int*)(smem + OFF_IDX);
  float*    sVal  = (float*)(smem + OFF_VAL);
  float*    sH2   = (float*)(smem + OFF_H2);     // [64][128]
  _Float16* sH2t  = (_Float16*)(smem + OFF_H2T); // [128][64]
  _Float16* sAt2  = (_Float16*)(smem + OFF_AT2); // [64][64]
  float*    sO2   = (float*)(smem + OFF_O2);     // [64][128]
  float*    sS2   = (float*)(smem + OFF_S2);
  float*    sT2   = (float*)(smem + OFF_T2);
  float*    sMean = (float*)(smem + OFF_MEAN);
  float*    sY1   = (float*)(smem + OFF_Y1);

  const int tid  = threadIdx.x;
  const int wave = tid >> 5;
  const int lane = tid & 31;
  const int b    = blockIdx.x;
  const float* xb = x + (size_t)b * NODE * NODE;

  // ---- P0: TDM-stage x_b [112x112] f32 directly into LDS -------------------
  // Generic->LDS: low 32 bits of a generic LDS pointer are the LDS byte offset
  // (flat aperture rule). One descriptor issued by wave 0; EXEC is ignored by
  // tensor ops, TENSORcnt tracks completion on the issuing wave.
  if (wave == 0) {
    tdm_load2d_f32_to_lds(xb, (uint32_t)(uintptr_t)(smem + OFF_X), NODE, NODE);
  }
  __builtin_amdgcn_s_wait_tensorcnt(0);   // no-op for waves with TENSORcnt==0
  __syncthreads();

  // ---- P1: att rows = expE * x, row-normalized, stored f16 k-padded --------
  if (tid < NODE) {
    const float* er = expE + tid * NODE;
    const float* xr = sX + tid * NODE;
    float rs = 0.0f;
    for (int j = 0; j < NODE; ++j) rs += er[j] * xr[j];
    const float inv = 1.0f / rs;
    _Float16* ar = sA + tid * 128;
    for (int j = 0; j < NODE; ++j) ar[j] = (_Float16)(er[j] * xr[j] * inv);
    for (int j = NODE; j < 128; ++j) ar[j] = (_Float16)0.0f;
  }
  __syncthreads();

  // ---- P2: GEMM1  h1 = relu(att @ H + H)   [112x256], K=128 (padded) ------
  for (int t = wave; t < 7 * 16; t += 8) {
    const int mt = t % 7, nt = t / 7;
    v8f acc = {};
#pragma unroll
    for (int kk = 0; kk < 4; ++kk) {
      v16h a = load_frag_A(sA, 64, mt, kk, lane);
      v16h bm = load_frag_Bt(Hf16t, 64, nt, kk, lane);
      acc = __builtin_amdgcn_wmma_f32_16x16x32_f16(false, a, false, bm,
                                                   (short)0, acc, false, false);
    }
    const int n = nt * 16 + (lane & 15);
    const float* hres = Hf32t + n * NODE;
    const int mBase = mt * 16 + ((lane >> 4) << 3);
#pragma unroll
    for (int r = 0; r < 8; ++r) {
      const int m = mBase + r;
      sH1[m * L1D + n] = (_Float16)fmaxf(acc[r] + hres[m], 0.0f);
    }
  }
  __syncthreads();

  // ---- P3: pool1 scores + exact top-56 (jax.lax.top_k semantics) ----------
  if (tid < NODE) {
    float acc = bpool1[0];
    const _Float16* hr = sH1 + tid * L1D;
    for (int f = 0; f < L1D; ++f) acc += (float)hr[f] * Wpool1[f];
    sSc[tid] = sigmoidf_(acc);
  }
  __syncthreads();
  if (tid < NODE) {
    const int r = topk_rank(sSc, NODE, tid);
    if (r < KP1) { sIdx[r] = tid; sVal[r] = sSc[tid]; }
  }
  __syncthreads();

  // ---- P4: gather hpA (f16, scaled) and a1 (normalized adj) ---------------
  for (int e = tid; e < 64 * L1D; e += 256) {
    const int r = e >> 8, f = e & 255;
    _Float16 v = (_Float16)0.0f;
    if (r < KP1) v = (_Float16)((float)sH1[sIdx[r] * L1D + f] * sVal[r]);
    sA[e] = v;  // hpA overlays attA (dead)
  }
  for (int e = tid; e < KP1 * KP1; e += 256) {
    const int r = e / KP1, c = e % KP1;
    sA1[e] = sX[sIdx[r] * NODE + sIdx[c]];
  }
  __syncthreads();
  if (tid < KP1) {
    float* ar = sA1 + tid * KP1;
    float rs = 0.0f;
    for (int c = 0; c < KP1; ++c) rs += ar[c];
    const float inv = 1.0f / rs;
    for (int c = 0; c < KP1; ++c) ar[c] *= inv;
  }
  __syncthreads();

  // ---- P5: GEMM2  h2 = hp @ Wp2^T + bp2   [64x128], K=256 -----------------
  for (int t = wave; t < 4 * 8; t += 8) {
    const int mt = t & 3, nt = t >> 2;
    v8f acc = {};
#pragma unroll
    for (int kk = 0; kk < 8; ++kk) {
      v16h a = load_frag_A(sA, 128, mt, kk, lane);
      v16h bm = load_frag_Bt(Wp2h, 128, nt, kk, lane);
      acc = __builtin_amdgcn_wmma_f32_16x16x32_f16(false, a, false, bm,
                                                   (short)0, acc, false, false);
    }
    const int n = nt * 16 + (lane & 15);
    const float bias = bp2[n];
    const int mBase = mt * 16 + ((lane >> 4) << 3);
#pragma unroll
    for (int r = 0; r < 8; ++r) sH2[(mBase + r) * L2D + n] = acc[r] + bias;
  }
  __syncthreads();

  // ---- P6: GAT2 attention logits + att2 (f16) + h2^T (f16) ----------------
  if (tid < KP1) {
    float s = 0.0f, tt = 0.0f;
    const float* hr = sH2 + tid * L2D;
    for (int f = 0; f < L2D; ++f) { s += hr[f] * wa2[f]; tt += hr[f] * wb2[f]; }
    sS2[tid] = s; sT2[tid] = tt;
  }
  __syncthreads();
  const float b2 = ba2[0];
  if (tid < 64) {
    _Float16* ar = sAt2 + tid * 64;
    if (tid < KP1) {
      const float si = sS2[tid];
      const float* a1r = sA1 + tid * KP1;
      float rs = 0.0f;
      for (int j = 0; j < KP1; ++j)
        rs += __expf(fmaxf(si + sT2[j] + b2, 0.0f)) * a1r[j];
      const float inv = 1.0f / rs;
      for (int j = 0; j < 64; ++j)
        ar[j] = (j < KP1)
            ? (_Float16)(__expf(fmaxf(si + sT2[j] + b2, 0.0f)) * a1r[j] * inv)
            : (_Float16)0.0f;
    } else {
      for (int j = 0; j < 64; ++j) ar[j] = (_Float16)0.0f;
    }
  }
  for (int e = tid; e < L2D * 64; e += 256) {
    const int n = e >> 6, k = e & 63;
    sH2t[e] = (k < KP1) ? (_Float16)sH2[k * L2D + n] : (_Float16)0.0f;
  }
  __syncthreads();

  // ---- P7: GEMM3  out2 = relu(att2 @ h2 + h2)   [64x128], K=64 ------------
  for (int t = wave; t < 4 * 8; t += 8) {
    const int mt = t & 3, nt = t >> 2;
    v8f acc = {};
#pragma unroll
    for (int kk = 0; kk < 2; ++kk) {
      v16h a = load_frag_A(sAt2, 32, mt, kk, lane);
      v16h bm = load_frag_Bt(sH2t, 32, nt, kk, lane);
      acc = __builtin_amdgcn_wmma_f32_16x16x32_f16(false, a, false, bm,
                                                   (short)0, acc, false, false);
    }
    const int n = nt * 16 + (lane & 15);
    const int mBase = mt * 16 + ((lane >> 4) << 3);
#pragma unroll
    for (int r = 0; r < 8; ++r) {
      const int m = mBase + r;
      if (m < KP1) sO2[m * L2D + n] = fmaxf(acc[r] + sH2[m * L2D + n], 0.0f);
    }
  }
  __syncthreads();

  // ---- P8: pool2 scores + top-28 ------------------------------------------
  if (tid < KP1) {
    float acc = bpool2[0];
    const float* hr = sO2 + tid * L2D;
    for (int f = 0; f < L2D; ++f) acc += hr[f] * Wpool2[f];
    sSc[tid] = sigmoidf_(acc);
  }
  __syncthreads();
  if (tid < KP1) {
    const int r = topk_rank(sSc, KP1, tid);
    if (r < KP2) { sIdx[r] = tid; sVal[r] = sSc[tid]; }
  }
  __syncthreads();

  // ---- P9: mean over selected (scaled) rows -------------------------------
  if (tid < L2D) {
    float acc = 0.0f;
    for (int r = 0; r < KP2; ++r) acc += sO2[sIdx[r] * L2D + tid] * sVal[r];
    sMean[tid] = acc * (1.0f / (float)KP2);
  }
  __syncthreads();

  // ---- P10: FC head --------------------------------------------------------
  if (tid < L3D) {
    float acc = bfc1[tid];
    const float* wr = Wfc1 + tid * L2D;
    for (int f = 0; f < L2D; ++f) acc += sMean[f] * wr[f];
    sY1[tid] = fmaxf(acc, 0.0f);
  }
  __syncthreads();
  if (tid == 0) {
    float acc = bfc2[0];
    for (int j = 0; j < L3D; ++j) acc += sY1[j] * Wfc2[j];
    out[b] = acc;
  }
}

// ---------------------------------------------------------------------------
extern "C" void kernel_launch(void* const* d_in, const int* in_sizes, int n_in,
                              void* d_out, int out_size, void* d_ws, size_t ws_size,
                              hipStream_t stream) {
  (void)in_sizes; (void)n_in; (void)out_size; (void)ws_size;

  const float* x      = (const float*)d_in[0];
  const float* Wp1    = (const float*)d_in[1];
  const float* bp1    = (const float*)d_in[2];
  const float* wa1    = (const float*)d_in[3];
  const float* wb1    = (const float*)d_in[4];
  const float* ba1    = (const float*)d_in[5];
  const float* Wpool1 = (const float*)d_in[6];
  const float* bpool1 = (const float*)d_in[7];
  const float* Wp2    = (const float*)d_in[8];
  const float* bp2    = (const float*)d_in[9];
  const float* wa2    = (const float*)d_in[10];
  const float* wb2    = (const float*)d_in[11];
  const float* ba2    = (const float*)d_in[12];
  const float* Wpool2 = (const float*)d_in[13];
  const float* bpool2 = (const float*)d_in[14];
  const float* Wfc1   = (const float*)d_in[15];
  const float* bfc1   = (const float*)d_in[16];
  const float* Wfc2   = (const float*)d_in[17];
  const float* bfc2   = (const float*)d_in[18];

  float*    Hf32t = (float*)((char*)d_ws + WS_HF32T);
  _Float16* Hf16t = (_Float16*)((char*)d_ws + WS_HF16T);
  float*    expE  = (float*)((char*)d_ws + WS_EXPE);
  _Float16* Wp2h  = (_Float16*)((char*)d_ws + WS_WP2H);

  gnn_setup<<<1, 256, 0, stream>>>(Wp1, bp1, wa1, wb1, ba1, Wp2,
                                   Hf32t, Hf16t, expE, Wp2h);
  gnn_main<<<BATCH, 256, 0, stream>>>(x, expE, Hf16t, Hf32t,
                                      Wpool1, bpool1, Wp2h, bp2, wa2, wb2, ba2,
                                      Wpool2, bpool2, Wfc1, bfc1, Wfc2, bfc2,
                                      (float*)d_out);
}
